// CausalAttention_8083128451355
// MI455X (gfx1250) — matrix operand order repaired and checked by reading, not compiled, not run on hardware
//
#include <hip/hip_runtime.h>
#include <hip/hip_bf16.h>

typedef _Float16 half_t;
typedef __attribute__((ext_vector_type(16))) _Float16 v16h;
typedef __attribute__((ext_vector_type(8)))  _Float16 v8h;
typedef __attribute__((ext_vector_type(4)))  _Float16 v4h;
typedef __attribute__((ext_vector_type(8)))  float    v8f;
typedef __attribute__((ext_vector_type(4)))  float    v4f;
typedef __attribute__((ext_vector_type(4)))  int      v4i;

#define SEQ_T   4096
#define DMODEL  1024
#define N3D     3072
#define NHEAD   16
#define HSZ     64

#if defined(__has_builtin)
#if __has_builtin(__builtin_amdgcn_global_load_async_to_lds_b128)
#define USE_ASYNC_LDS 1
#endif
#endif
#ifndef USE_ASYNC_LDS
#define USE_ASYNC_LDS 0
#endif

#define GLOBAL_AS __attribute__((address_space(1)))
#define LDS_AS    __attribute__((address_space(3)))

__device__ __forceinline__ void wait_async0() {
#if USE_ASYNC_LDS
  asm volatile("s_wait_asynccnt 0x0" ::: "memory");
#endif
}

__device__ __forceinline__ v8f wmma_f16(v16h a, v16h b, v8f c) {
  // D = A(16x32 f16) * B(32x16 f16) + C(16x16 f32)
  return __builtin_amdgcn_wmma_f32_16x16x32_f16(false, a, false, b, (short)0, c,
                                                false, false);
}

// A-matrix 16x32 f16 fragment from LDS (row-major, row stride ld halfs).
// ISA 7.12.2: lane L: M=L&15, hs=L>>4: elems 0..7 -> K=8*hs+e,
// elems 8..15 -> K=16+8*hs+(e-8). Two contiguous 16B ds reads per lane.
__device__ __forceinline__ v16h load_a_frag(const half_t* base, int ld,
                                            int mBase, int kBase, int lane) {
  const int m  = mBase + (lane & 15);
  const int hs = lane >> 4;
  const half_t* row = base + m * ld + kBase;
  v8h lo = *(const v8h*)(row + 8 * hs);
  v8h hi = *(const v8h*)(row + 16 + 8 * hs);
  v16h a;
#pragma unroll
  for (int i = 0; i < 8; ++i) { a[i] = lo[i]; a[i + 8] = hi[i]; }
  return a;
}

// B-matrix 32x16 f16 fragment read from a TRANSPOSED LDS tile Bt[n][k]
// (row stride ld halfs). Lane L: N = L&15, elems 0..7 -> K=8*hs+e,
// elems 8..15 -> K=16+8*hs+(e-8).
__device__ __forceinline__ v16h load_bt_frag(const half_t* base, int ld,
                                             int nBase, int kBase, int lane) {
  const int n  = nBase + (lane & 15);
  const int hs = lane >> 4;
  const half_t* row = base + n * ld + kBase;
  v8h lo = *(const v8h*)(row + 8 * hs);
  v8h hi = *(const v8h*)(row + 16 + 8 * hs);
  v16h b;
#pragma unroll
  for (int i = 0; i < 8; ++i) { b[i] = lo[i]; b[i + 8] = hi[i]; }
  return b;
}

// ---------------------------------------------------------------------------
// Kernel 1: qkv = x @ Wqkv + bqkv  (f16 WMMA, f32 accum).
// Output scatter-stored f16 in layout qkvh[s][h][t][d] matching the
// reference's raw reshape(3,B,T,HS,NH).permute(0,1,4,2,3).
// Grid: (N3D/128, T/128), block 256 (8 waves, each 64x32 of 128x128 tile).
// K-step 64 (two WMMA K-slices per staged tile).
// ---------------------------------------------------------------------------
__global__ __launch_bounds__(256) void qkv_gemm_kernel(
    const float* __restrict__ x, const float* __restrict__ W,
    const float* __restrict__ bias, half_t* __restrict__ qkvh) {
  __shared__ half_t As[128 * 72];  // [m][k], k pad 64->72 (144B rows)
  __shared__ half_t Bs[128 * 72];  // transposed: [n][k]

  const int tid  = threadIdx.x;
  const int wave = tid >> 5;
  const int lane = tid & 31;
  const int m0 = blockIdx.y * 128;
  const int n0 = blockIdx.x * 128;
  const int wm = wave & 1;   // 2 M-halves of 64
  const int wn = wave >> 1;  // 4 N-quarters of 32

  const v8f z8 = {0.f, 0.f, 0.f, 0.f, 0.f, 0.f, 0.f, 0.f};
  v8f acc[4][2];
#pragma unroll
  for (int i = 0; i < 4; ++i)
#pragma unroll
    for (int j = 0; j < 2; ++j) acc[i][j] = z8;

  for (int k0 = 0; k0 < DMODEL; k0 += 64) {
    // stage A (128x64): vectorized f32x4 loads, packed f16x4 LDS stores
#pragma unroll
    for (int i = 0; i < 8; ++i) {
      int e4 = i * 256 + tid;          // 2048 float4
      int r = e4 >> 4, c4 = e4 & 15;   // 16 float4 per row
      v4f f = *(const v4f*)(x + (m0 + r) * DMODEL + k0 + c4 * 4);
      v4h hv;
#pragma unroll
      for (int j = 0; j < 4; ++j) hv[j] = (half_t)f[j];
      *(v4h*)(&As[r * 72 + c4 * 4]) = hv;
    }
    // stage B (64x128) transposed into Bt[n][k]
#pragma unroll
    for (int i = 0; i < 8; ++i) {
      int e4 = i * 256 + tid;            // 2048 float4
      int kr = e4 >> 5, nc4 = e4 & 31;   // 32 float4 per k-row
      v4f f = *(const v4f*)(W + (k0 + kr) * N3D + n0 + nc4 * 4);
#pragma unroll
      for (int j = 0; j < 4; ++j)
        Bs[(nc4 * 4 + j) * 72 + kr] = (half_t)f[j];
    }
    __syncthreads();

    v16h bf[2][2];
#pragma unroll
    for (int nt = 0; nt < 2; ++nt) {
      bf[nt][0] = load_bt_frag(Bs, 72, wn * 32 + nt * 16, 0, lane);
      bf[nt][1] = load_bt_frag(Bs, 72, wn * 32 + nt * 16, 32, lane);
    }
#pragma unroll
    for (int mt = 0; mt < 4; ++mt) {
      v16h a0 = load_a_frag(As, 72, wm * 64 + mt * 16, 0, lane);
      v16h a1 = load_a_frag(As, 72, wm * 64 + mt * 16, 32, lane);
#pragma unroll
      for (int nt = 0; nt < 2; ++nt) {
        acc[mt][nt] = wmma_f16(a0, bf[nt][0], acc[mt][nt]);
        acc[mt][nt] = wmma_f16(a1, bf[nt][1], acc[mt][nt]);
      }
    }
    __syncthreads();
  }

  // Epilogue: fp32 bias add + scatter into rearranged f16 layout.
  const int nloc = lane & 15, hsv = lane >> 4;
#pragma unroll
  for (int mt = 0; mt < 4; ++mt) {
#pragma unroll
    for (int nt = 0; nt < 2; ++nt) {
      const int c = n0 + wn * 32 + nt * 16 + nloc;
      const float bv = bias[c];
#pragma unroll
      for (int r = 0; r < 8; ++r) {
        const int m = m0 + wm * 64 + mt * 16 + r + 8 * hsv;
        const unsigned f   = (unsigned)m * N3D + c;  // raw flat index
        const unsigned s   = f >> 22;                // / (T*D)
        const unsigned rem = f & 4194303u;
        const unsigned t2  = rem >> 10;              // / (HS*NH)
        const unsigned rr  = rem & 1023u;
        const unsigned d   = rr >> 4;                // / NH
        const unsigned h   = rr & 15u;
        qkvh[(((s * 16 + h) * 4096 + t2) << 6) + d] =
            (half_t)(acc[mt][nt][r] + bv);
      }
    }
  }
}

// ---------------------------------------------------------------------------
// Kernel 2: flash-style causal attention per head.
// Grid: (T/128, NH), block 256 (8 waves; each wave owns 16 query rows).
// ---------------------------------------------------------------------------
__global__ __launch_bounds__(256) void attn_kernel(
    const half_t* __restrict__ qkvh, float* __restrict__ out) {
  __shared__ half_t Qs[128 * 72];    // [q][hs]    (pad 64->72)
  __shared__ half_t Ks[64 * 72];     // [key][hs] == Bt layout for Q*K^T
  __shared__ half_t Vt[64 * 72];     // [hs][key] == Bt layout for P*V
  __shared__ half_t Ps[8][16 * 72];  // per-wave P staging (C-layout -> A-frag)

  const int tid  = threadIdx.x;
  const int wave = tid >> 5;
  const int lane = tid & 31;
  const int h    = blockIdx.y;
  const int q0   = blockIdx.x * 128;
  const int mB   = wave * 16;
  const int nloc = lane & 15, hsv = lane >> 4;

  const half_t* Q = qkvh + (size_t)(0  + h) * SEQ_T * HSZ;
  const half_t* K = qkvh + (size_t)(16 + h) * SEQ_T * HSZ;
  const half_t* V = qkvh + (size_t)(32 + h) * SEQ_T * HSZ;

  // stage Q tile (128x64): async global->LDS if available, else 16B copies
#pragma unroll
  for (int i = 0; i < 4; ++i) {
    int ve = i * 256 + tid;  // 1024 v8h
    int r = ve >> 3, c8 = ve & 7;
#if USE_ASYNC_LDS
    __builtin_amdgcn_global_load_async_to_lds_b128(
        (GLOBAL_AS v4i*)(Q + (q0 + r) * HSZ + c8 * 8),
        (LDS_AS v4i*)(&Qs[r * 72 + c8 * 8]), 0, 0);
#else
    *(v8h*)(&Qs[r * 72 + c8 * 8]) = *(const v8h*)(Q + (q0 + r) * HSZ + c8 * 8);
#endif
  }

  const v8f z8 = {0.f, 0.f, 0.f, 0.f, 0.f, 0.f, 0.f, 0.f};
  v8f o[4];
#pragma unroll
  for (int i = 0; i < 4; ++i) o[i] = z8;
  float mrow[8], lrow[8];
#pragma unroll
  for (int r = 0; r < 8; ++r) { mrow[r] = -3e38f; lrow[r] = 0.f; }

  const int nkt = 2 * blockIdx.x + 2;  // causal: key tiles [0, 2*qt+1]
  for (int kt = 0; kt < nkt; ++kt) {
    const int k0 = kt * 64;
    // stage K tile [key][hs]
#pragma unroll
    for (int i = 0; i < 2; ++i) {
      int ve = i * 256 + tid;  // 512 v8h
      int r = ve >> 3, c8 = ve & 7;
#if USE_ASYNC_LDS
      __builtin_amdgcn_global_load_async_to_lds_b128(
          (GLOBAL_AS v4i*)(K + (k0 + r) * HSZ + c8 * 8),
          (LDS_AS v4i*)(&Ks[r * 72 + c8 * 8]), 0, 0);
#else
      *(v8h*)(&Ks[r * 72 + c8 * 8]) =
          *(const v8h*)(K + (k0 + r) * HSZ + c8 * 8);
#endif
    }
    // stage V transposed: Vt[hs][key] (coalesced global reads along hs)
#pragma unroll
    for (int i = 0; i < 16; ++i) {
      int e = i * 256 + tid;
      int r = e >> 6, c = e & 63;  // r=key, c=hs
      Vt[c * 72 + r] = V[(k0 + r) * HSZ + c];
    }
    // prefetch next K/V tiles into cache while we compute
    if (kt + 1 < nkt) {
      __builtin_prefetch((const void*)(K + (k0 + 64) * HSZ + tid * 16), 0, 3);
      __builtin_prefetch((const void*)(V + (k0 + 64) * HSZ + tid * 16), 0, 3);
    }
    wait_async0();
    __syncthreads();

    // S = Q_wave(16x64) @ K^T(64x64): 4 N-tiles x 2 K-slices
    v8f s[4];
#pragma unroll
    for (int i = 0; i < 4; ++i) s[i] = z8;
    v16h a0 = load_a_frag(Qs, 72, mB, 0, lane);
    v16h a1 = load_a_frag(Qs, 72, mB, 32, lane);
#pragma unroll
    for (int nt = 0; nt < 4; ++nt) {
      v16h b0 = load_bt_frag(Ks, 72, nt * 16, 0, lane);
      v16h b1 = load_bt_frag(Ks, 72, nt * 16, 32, lane);
      s[nt] = wmma_f16(a0, b0, s[nt]);
      s[nt] = wmma_f16(a1, b1, s[nt]);
    }

    // Online softmax. C-layout: row M = r + 8*hsv, col N = nt*16 + nloc.
    const bool needMask = (k0 + 63 > q0 + mB);
#pragma unroll
    for (int r = 0; r < 8; ++r) {
      const int qg = q0 + mB + r + 8 * hsv;
      float mx = mrow[r];
#pragma unroll
      for (int nt = 0; nt < 4; ++nt) {
        float vv = s[nt][r] * 0.125f;  // HS^-0.5
        if (needMask) {
          const int kg = k0 + nt * 16 + nloc;
          if (kg > qg) vv = -3e38f;
        }
        s[nt][r] = vv;
        mx = fmaxf(mx, vv);
      }
#pragma unroll
      for (int off = 1; off < 16; off <<= 1)
        mx = fmaxf(mx, __shfl_xor(mx, off, 16));
      const float sc = __expf(mrow[r] - mx);
      float rs = 0.f;
#pragma unroll
      for (int nt = 0; nt < 4; ++nt) {
        const float p = __expf(s[nt][r] - mx);
        s[nt][r] = p;
        rs += p;
      }
#pragma unroll
      for (int off = 1; off < 16; off <<= 1) rs += __shfl_xor(rs, off, 16);
      lrow[r] = lrow[r] * sc + rs;
      mrow[r] = mx;
#pragma unroll
      for (int nt = 0; nt < 4; ++nt) o[nt][r] *= sc;
    }

    // Stage P (f16) through per-wave LDS to convert C-layout -> A-fragment.
    half_t* P = &Ps[wave][0];
#pragma unroll
    for (int r = 0; r < 8; ++r)
#pragma unroll
      for (int nt = 0; nt < 4; ++nt)
        P[(r + 8 * hsv) * 72 + nt * 16 + nloc] = (half_t)s[nt][r];

    v16h p0 = load_a_frag(P, 72, 0, 0, lane);
    v16h p1 = load_a_frag(P, 72, 0, 32, lane);
#pragma unroll
    for (int nt = 0; nt < 4; ++nt) {
      v16h b0 = load_bt_frag(Vt, 72, nt * 16, 0, lane);
      v16h b1 = load_bt_frag(Vt, 72, nt * 16, 32, lane);
      o[nt] = wmma_f16(p0, b0, o[nt]);
      o[nt] = wmma_f16(p1, b1, o[nt]);
    }
    __syncthreads();
  }

  // out[t][h*64+d] = O / l
#pragma unroll
  for (int r = 0; r < 8; ++r) {
    const int tq = q0 + mB + r + 8 * hsv;
    const float inv = 1.0f / lrow[r];
#pragma unroll
    for (int nt = 0; nt < 4; ++nt) {
      const int d = nt * 16 + nloc;
      out[tq * DMODEL + h * HSZ + d] = o[nt][r] * inv;
    }
  }
}

extern "C" void kernel_launch(void* const* d_in, const int* in_sizes, int n_in,
                              void* d_out, int out_size, void* d_ws,
                              size_t ws_size, hipStream_t stream) {
  (void)in_sizes; (void)n_in; (void)out_size; (void)ws_size;
  const float* x    = (const float*)d_in[0];
  const float* Wqkv = (const float*)d_in[1];
  const float* bqkv = (const float*)d_in[2];
  float* out   = (float*)d_out;
  half_t* qkvh = (half_t*)d_ws;  // 3*16*4096*64 f16 = 25.2 MB

  dim3 g1(N3D / 128, SEQ_T / 128);
  qkv_gemm_kernel<<<g1, dim3(256), 0, stream>>>(x, Wqkv, bqkv, qkvh);

  dim3 g2(SEQ_T / 128, NHEAD);
  attn_kernel<<<g2, dim3(256), 0, stream>>>(qkvh, out);
}
